// MultiHeadAttentionModel_41953240547584
// MI455X (gfx1250) — matrix operand order repaired and checked by reading, not compiled, not run on hardware
//
#include <hip/hip_runtime.h>

#define DEV __device__ __forceinline__

typedef __bf16 bf16;
typedef __attribute__((ext_vector_type(16))) __bf16 v16bf;
typedef __attribute__((ext_vector_type(8)))  __bf16 v8bf;
typedef __attribute__((ext_vector_type(4)))  __bf16 v4bf;
typedef __attribute__((ext_vector_type(8)))  float  v8f;
typedef __attribute__((ext_vector_type(4)))  int    v4i;

constexpr int NB  = 2;
constexpr int NT  = 2048;
constexpr int NH  = 16;
constexpr int NHS = 64;
constexpr int NC  = 1024;          // NH * NHS
constexpr int NV  = 50257;
constexpr int VPAD = 50304;        // NV rounded up to multiple of 64
constexpr int TT  = NT / 16;       // 128 row-tiles per (b,h)

// ---- gfx1250 async load-to-LDS (guarded: falls back to load + ds_store) ----
#if defined(__gfx1250__) && __has_builtin(__builtin_amdgcn_global_load_async_to_lds_b128)
#define HAS_ASYNC 1
#else
#define HAS_ASYNC 0
#endif

template <int N> DEV void wait_async() {
#if HAS_ASYNC
#if __has_builtin(__builtin_amdgcn_s_wait_asynccnt)
  __builtin_amdgcn_s_wait_asynccnt(N);
#else
  asm volatile("s_wait_asynccnt %0" ::"i"(N) : "memory");
#endif
#endif
}

DEV void async_copy16(const bf16* src, bf16* lds_dst) {
#if HAS_ASYNC
  // Probe-derived signature: (int4 addrspace(1)*, int4 addrspace(3)*, imm, imm)
  __builtin_amdgcn_global_load_async_to_lds_b128(
      (__attribute__((address_space(1))) v4i*)src,
      (__attribute__((address_space(3))) v4i*)lds_dst, 0, 0);
#else
  *(v8bf*)lds_dst = *(const v8bf*)src;
#endif
}

// ---------------------------------------------------------------- WMMA helpers
DEV v8f wmma_bf16(v16bf a, v16bf b, v8f c) {
  // D = A(16x32 bf16) * B(32x16 bf16) + C(16x16 f32)
  return __builtin_amdgcn_wmma_f32_16x16x32_bf16(false, a, false, b,
                                                 (short)0, c, false, false);
}

// A fragment: lane = 16*g + m owns row m; element e -> K = 16*(e/8) + 8*g + e%8
DEV v16bf load_a16(const bf16* row_k0, int g) {
  union { v16bf v; v8bf h[2]; } u;
  u.h[0] = *(const v8bf*)(row_k0 + 8 * g);
  u.h[1] = *(const v8bf*)(row_k0 + 16 + 8 * g);
  return u.v;
}

// B fragment: lane = 16*g + n owns column n; element e -> K = 16*(e/8) + 8*g + e%8
DEV v16bf load_b16(const bf16* row_k0, int g) {
  union { v16bf v; v8bf h[2]; } u;
  u.h[0] = *(const v8bf*)(row_k0 + 8 * g);
  u.h[1] = *(const v8bf*)(row_k0 + 16 + 8 * g);
  return u.v;
}

DEV v8bf pack8(v8f f) {
  v8bf r;
#pragma unroll
  for (int i = 0; i < 8; ++i) r[i] = (bf16)f[i];
  return r;
}

// ---------------------------------------------------------------- 1) embedding
__global__ __launch_bounds__(256) void embed_kernel(const int* __restrict__ idx,
                                                    const float* __restrict__ Wtok,
                                                    const float* __restrict__ Wpos,
                                                    bf16* __restrict__ xbf) {
  const int bt  = blockIdx.x;            // 0 .. NB*NT-1
  const int t   = bt % NT;
  const int tok = idx[bt];
  const int c   = threadIdx.x * 4;       // 256 threads * 4 = NC
  const float4 a = *(const float4*)(Wtok + (size_t)tok * NC + c);
  const float4 p = *(const float4*)(Wpos + (size_t)t   * NC + c);
  v4bf o;
  o[0] = (bf16)(a.x + p.x);
  o[1] = (bf16)(a.y + p.y);
  o[2] = (bf16)(a.z + p.z);
  o[3] = (bf16)(a.w + p.w);
  *(v4bf*)(xbf + (size_t)bt * NC + c) = o;
}

// ------------------------------------------------- 2) weight convert/transpose
__global__ __launch_bounds__(256) void conv_qkv_w(const float* __restrict__ Wq,
                                                  const float* __restrict__ Wk,
                                                  const float* __restrict__ Wv,
                                                  bf16* __restrict__ WqT,
                                                  bf16* __restrict__ WkT,
                                                  bf16* __restrict__ WvT) {
  size_t i = (size_t)blockIdx.x * blockDim.x + threadIdx.x; // (w,h,d,c), c fastest
  const int c = (int)(i % NC); size_t r = i / NC;
  const int d = (int)(r % NHS); r /= NHS;
  const int h = (int)(r % NH);
  const int w = (int)(r / NH);
  const float* src = (w == 0) ? Wq : (w == 1) ? Wk : Wv;
  bf16*        dst = (w == 0) ? WqT : (w == 1) ? WkT : WvT;
  dst[((size_t)h * NHS + d) * NC + c] = (bf16)src[((size_t)h * NC + c) * NHS + d];
}

__global__ __launch_bounds__(256) void conv_wl(const float* __restrict__ Wl,
                                               bf16* __restrict__ WlT) {
  size_t i = (size_t)blockIdx.x * blockDim.x + threadIdx.x; // (n, c), c fastest
  const int c = (int)(i % NC);
  const int n = (int)(i / NC);
  WlT[i] = (n < NV) ? (bf16)Wl[(size_t)c * NV + n] : (bf16)0.0f;
}

// --------------------------------------------------------- 3) QKV projections
// One wave per (w, b, h, 16-row tile); w selects q/k/v. 2-stage software
// pipeline, unrolled x2 so the cur<-nxt rotation becomes register renaming.
__global__ __launch_bounds__(128, 1) void qkv_kernel(
    const bf16* __restrict__ xbf,
    const bf16* __restrict__ WqT,
    const bf16* __restrict__ WkT,
    const bf16* __restrict__ WvT,
    bf16* __restrict__ q,
    bf16* __restrict__ k,
    bf16* __restrict__ vT) {
  const int wave = (int)((blockIdx.x * blockDim.x + threadIdx.x) >> 5);
  const int lane = (int)(threadIdx.x & 31);
  const int g = lane >> 4, ln = lane & 15;
  const int w = wave % 3; int rem = wave / 3;       // 0=q, 1=k, 2=v
  const int tt = rem % TT; rem /= TT;
  const int h = rem % NH, b = rem / NH;
  const int t0 = tt * 16;

  const bf16* Arow = xbf + (size_t)(b * NT + t0 + ln) * NC;   // A: lane row t0+ln
  const bf16* WT = (w == 0) ? WqT : (w == 1) ? WkT : WvT;
  const bf16* Wh = WT + (size_t)h * NHS * NC;

  v8f acc[4] = {};
  // ---- pipeline prologue: fragments for k0 = 0
  v16bf a_cur = load_a16(Arow, g);
  v16bf b_cur[4];
#pragma unroll
  for (int dg = 0; dg < 4; ++dg)
    b_cur[dg] = load_b16(Wh + (size_t)(dg * 16 + ln) * NC, g);

#pragma unroll 2
  for (int k0 = 32; k0 < NC; k0 += 32) {
    // issue next iteration's loads first
    v16bf a_nxt = load_a16(Arow + k0, g);
    v16bf b_nxt[4];
#pragma unroll
    for (int dg = 0; dg < 4; ++dg)
      b_nxt[dg] = load_b16(Wh + (size_t)(dg * 16 + ln) * NC + k0, g);
    // consume current fragments
#pragma unroll
    for (int dg = 0; dg < 4; ++dg)
      acc[dg] = wmma_bf16(a_cur, b_cur[dg], acc[dg]);
    a_cur = a_nxt;
#pragma unroll
    for (int dg = 0; dg < 4; ++dg) b_cur[dg] = b_nxt[dg];
  }
  // ---- pipeline epilogue
#pragma unroll
  for (int dg = 0; dg < 4; ++dg)
    acc[dg] = wmma_bf16(a_cur, b_cur[dg], acc[dg]);

  if (w == 2) {
    // v transposed [B,H,HS,T]: lane owns row d, 8 consecutive t -> b128 store
    const size_t headVT = (size_t)(b * NH + h) * NHS * NT;
#pragma unroll
    for (int dg = 0; dg < 4; ++dg)
      *(v8bf*)(vT + headVT + (size_t)(dg * 16 + ln) * NT + t0 + 8 * g) =
          pack8(acc[dg]);
  } else {
    bf16* dst = (w == 0) ? q : k;
    const size_t headQK = ((size_t)(b * NH + h) * NT + t0) * NHS;
#pragma unroll
    for (int dg = 0; dg < 4; ++dg)
#pragma unroll
      for (int rr = 0; rr < 8; ++rr)
        dst[headQK + (size_t)(rr + 8 * g) * NHS + dg * 16 + ln] =
            (bf16)acc[dg][rr];
  }
}

// ------------------------------------------------- 4) flash attention (causal)
__global__ __launch_bounds__(128, 1) void attn_kernel(
    const bf16* __restrict__ q,
    const bf16* __restrict__ k,
    const bf16* __restrict__ vT,
    bf16* __restrict__ att) {
  __shared__ __align__(16) bf16 lds[4][16 * 40];   // P transpose staging / wave
  const int wv   = (int)(threadIdx.x >> 5);
  const int wave = (int)(blockIdx.x * (blockDim.x >> 5)) + wv;
  const int lane = (int)(threadIdx.x & 31);
  const int g = lane >> 4, ln = lane & 15;
  const int tt = wave % TT; int rr_ = wave / TT;
  const int h = rr_ % NH, b = rr_ / NH;
  const int t0 = tt * 16;

  const size_t headQK = (size_t)(b * NH + h) * NT * NHS;
  const size_t headVT = (size_t)(b * NH + h) * NHS * NT;
  const bf16* kBase = k + headQK;

  // Q fragments (A): invariant over s-loop.
  const bf16* qrow = q + headQK + (size_t)(t0 + ln) * NHS;
  const v16bf aq0 = load_a16(qrow + 0, g);
  const v16bf aq1 = load_a16(qrow + 32, g);

  float m[8], l[8];
#pragma unroll
  for (int i = 0; i < 8; ++i) { m[i] = -1e30f; l[i] = 0.0f; }
  v8f oacc[4] = {};

  const int sEnd = t0 + 16;                 // valid s <= t <= t0+15

  // ---- pipeline prologue: K fragments for s0 = 0
  v16bf bkc[4];
  {
    const bf16* kr0 = kBase + (size_t)ln * NHS;
    const bf16* kr1 = kBase + (size_t)(16 + ln) * NHS;
    bkc[0] = load_b16(kr0, g);      bkc[1] = load_b16(kr0 + 32, g);
    bkc[2] = load_b16(kr1, g);      bkc[3] = load_b16(kr1 + 32, g);
  }

  for (int s0 = 0; s0 < sEnd; s0 += 32) {
    // ---- score WMMAs with current K fragments
    v8f sc0 = {}, sc1 = {};
    sc0 = wmma_bf16(aq0, bkc[0], sc0);
    sc0 = wmma_bf16(aq1, bkc[1], sc0);
    sc1 = wmma_bf16(aq0, bkc[2], sc1);
    sc1 = wmma_bf16(aq1, bkc[3], sc1);

    // ---- issue next-chunk K loads (clamped: last iter reloads current,
    // always in-bounds) and this chunk's V loads; both overlap the softmax.
    const int sn = (s0 + 32 < sEnd) ? s0 + 32 : s0;
    v16bf bkn[4];
    {
      const bf16* kr0 = kBase + (size_t)(sn + ln) * NHS;
      const bf16* kr1 = kBase + (size_t)(sn + 16 + ln) * NHS;
      bkn[0] = load_b16(kr0, g);    bkn[1] = load_b16(kr0 + 32, g);
      bkn[2] = load_b16(kr1, g);    bkn[3] = load_b16(kr1 + 32, g);
    }
    v16bf bv[4];
#pragma unroll
    for (int dg = 0; dg < 4; ++dg)
      bv[dg] = load_b16(vT + headVT + (size_t)(dg * 16 + ln) * NT + s0, g);

    // ---- causal mask (only diagonal chunks)
    if (s0 + 31 > t0) {
#pragma unroll
      for (int rr = 0; rr < 8; ++rr) {
        const int t = t0 + rr + 8 * g;
        if (s0 + ln      > t) sc0[rr] = -1e30f;
        if (s0 + 16 + ln > t) sc1[rr] = -1e30f;
      }
    }

    // ---- online softmax (row lives in one 16-lane half -> xor-shuffle reduce)
#pragma unroll
    for (int rr = 0; rr < 8; ++rr) {
      float mx = fmaxf(sc0[rr], sc1[rr]);
#pragma unroll
      for (int o = 8; o >= 1; o >>= 1) mx = fmaxf(mx, __shfl_xor(mx, o, 32));
      const float mn    = fmaxf(m[rr], mx);
      const float alpha = __expf(m[rr] - mn);
      const float p0 = __expf(sc0[rr] - mn);
      const float p1 = __expf(sc1[rr] - mn);
      float rs = p0 + p1;
#pragma unroll
      for (int o = 8; o >= 1; o >>= 1) rs += __shfl_xor(rs, o, 32);
      l[rr] = l[rr] * alpha + rs;
      m[rr] = mn;
#pragma unroll
      for (int dg = 0; dg < 4; ++dg) oacc[dg][rr] *= alpha;
      // stage P (C-layout -> LDS row-major, row stride 40 halves, 16B aligned)
      lds[wv][(rr + 8 * g) * 40 + ln]      = (bf16)p0;
      lds[wv][(rr + 8 * g) * 40 + 16 + ln] = (bf16)p1;
    }

    // ---- read P back as A fragment (16x32), then PV accumulate
    union { v16bf v; v8bf hh[2]; } ua;
    ua.hh[0] = *(const v8bf*)&lds[wv][ln * 40 + 8 * g];
    ua.hh[1] = *(const v8bf*)&lds[wv][ln * 40 + 16 + 8 * g];
#pragma unroll
    for (int dg = 0; dg < 4; ++dg)
      oacc[dg] = wmma_bf16(ua.v, bv[dg], oacc[dg]);

#pragma unroll
    for (int i = 0; i < 4; ++i) bkc[i] = bkn[i];
  }

  // ---- epilogue: att[b, t, h*64 + d] (bf16), normalized
#pragma unroll
  for (int dg = 0; dg < 4; ++dg)
#pragma unroll
    for (int rr = 0; rr < 8; ++rr)
      att[(size_t)(b * NT + t0 + rr + 8 * g) * NC + h * NHS + dg * 16 + ln] =
          (bf16)(oacc[dg][rr] / l[rr]);
}

// ------------------------------------------------------------ 5) logits GEMM
// M = 4096, N = VPAD, K = NC. Block = 4 waves sharing one 64-col Wl panel
// staged through LDS (double-buffered, async-to-LDS); each wave owns a 16-row
// tile => Wl L2 traffic / 4. Unroll x2 makes the buffer index compile-time.
__global__ __launch_bounds__(128, 1) void logits_kernel(
    const bf16* __restrict__ att,
    const bf16* __restrict__ WlT,
    const float* __restrict__ bl,
    float* __restrict__ out) {
  __shared__ __align__(16) bf16 bs[2][64 * 40];    // 64 cols x 32 k, stride 40
  const int tid  = (int)threadIdx.x;
  const int wv   = tid >> 5;
  const int lane = tid & 31;
  const int g = lane >> 4, ln = lane & 15;
  constexpr int NCT = VPAD / 64;                   // 786 col tiles
  const int ct = (int)(blockIdx.x % NCT);
  const int rb = (int)(blockIdx.x / NCT);          // 0..63 row blocks of 64
  const int col0 = ct * 64;
  const int row0 = rb * 64 + wv * 16;

  const bf16* Arow = att + (size_t)(row0 + ln) * NC;

  // Stage one 64x32 bf16 panel of WlT: 256 16B chunks, 2 per thread.
  auto stage = [&](int buf, int k0) {
#pragma unroll
    for (int c2 = 0; c2 < 2; ++c2) {
      const int chunk = tid + c2 * 128;
      const int col = chunk >> 2, part = chunk & 3;
      async_copy16(WlT + (size_t)(col0 + col) * NC + k0 + part * 8,
                   &bs[buf][col * 40 + part * 8]);
    }
  };

  stage(0, 0);
  v8f acc[4] = {};
#pragma unroll 2
  for (int k0 = 0; k0 < NC; k0 += 32) {
    const int buf = (k0 >> 5) & 1;
    const bool hasNext = (k0 + 32) < NC;
    if (hasNext) { stage(buf ^ 1, k0 + 32); wait_async<2>(); }
    else         { wait_async<0>(); }
    __builtin_prefetch(Arow + k0 + 256, 0, 0);     // global_prefetch_b8
    const v16bf a = load_a16(Arow + k0, g);
    __syncthreads();                               // panel `buf` fully staged
    v16bf bfr[4];
#pragma unroll
    for (int cg = 0; cg < 4; ++cg) {
      const bf16* Brow = &bs[buf][(cg * 16 + ln) * 40];
      union { v16bf v; v8bf hh[2]; } u;
      u.hh[0] = *(const v8bf*)(Brow + 8 * g);
      u.hh[1] = *(const v8bf*)(Brow + 16 + 8 * g);
      bfr[cg] = u.v;
    }
#pragma unroll
    for (int cg = 0; cg < 4; ++cg)
      acc[cg] = wmma_bf16(a, bfr[cg], acc[cg]);
    __syncthreads();                               // done reading `buf`
  }

#pragma unroll
  for (int cg = 0; cg < 4; ++cg) {
    const int col = col0 + cg * 16 + ln;
    if (col < NV) {
      const float bias = bl[col];
#pragma unroll
      for (int rr = 0; rr < 8; ++rr)
        out[(size_t)(row0 + rr + 8 * g) * NV + col] = acc[cg][rr] + bias;
    }
  }
}

// ---------------------------------------------------------------------- launch
extern "C" void kernel_launch(void* const* d_in, const int* in_sizes, int n_in,
                              void* d_out, int out_size, void* d_ws, size_t ws_size,
                              hipStream_t stream) {
  const int*   idx  = (const int*)  d_in[0];
  const float* Wtok = (const float*)d_in[1];
  const float* Wpos = (const float*)d_in[2];
  const float* Wq   = (const float*)d_in[3];
  const float* Wk   = (const float*)d_in[4];
  const float* Wv   = (const float*)d_in[5];
  const float* Wl   = (const float*)d_in[6];
  const float* bl   = (const float*)d_in[7];
  float* out = (float*)d_out;

  char* ws = (char*)d_ws;
  size_t off = 0;
  auto alloc = [&](size_t bytes) -> void* {
    void* p = ws + off;
    off += (bytes + 255) & ~(size_t)255;
    return p;
  };
  bf16* xbf = (bf16*)alloc((size_t)NB * NT * NC * 2);
  bf16* WqT = (bf16*)alloc((size_t)NH * NHS * NC * 2);
  bf16* WkT = (bf16*)alloc((size_t)NH * NHS * NC * 2);
  bf16* WvT = (bf16*)alloc((size_t)NH * NHS * NC * 2);
  bf16* qb  = (bf16*)alloc((size_t)NB * NH * NT * NHS * 2);
  bf16* kb  = (bf16*)alloc((size_t)NB * NH * NT * NHS * 2);
  bf16* vTb = (bf16*)alloc((size_t)NB * NH * NHS * NT * 2);
  bf16* att = (bf16*)alloc((size_t)NB * NT * NC * 2);
  bf16* WlT = (bf16*)alloc((size_t)VPAD * NC * 2);

  embed_kernel<<<NB * NT, 256, 0, stream>>>(idx, Wtok, Wpos, xbf);
  conv_qkv_w<<<(3 * NH * NHS * NC) / 256, 256, 0, stream>>>(Wq, Wk, Wv,
                                                            WqT, WkT, WvT);
  conv_wl<<<((size_t)VPAD * NC) / 256, 256, 0, stream>>>(Wl, WlT);
  {
    const int waves = 3 * NB * NH * TT;       // 12288 (one per q/k/v tile)
    qkv_kernel<<<waves / 4, 128, 0, stream>>>(xbf, WqT, WkT, WvT, qb, kb, vTb);
  }
  {
    const int waves = NB * NH * TT;           // 4096
    attn_kernel<<<waves / 4, 128, 0, stream>>>(qb, kb, vTb, att);
  }
  {
    const int blocks = (NB * NT / 64) * (VPAD / 64);  // 64 * 786 = 50304
    logits_kernel<<<blocks, 128, 0, stream>>>(att, WlT, bl, out);
  }
}
